// theozyme_layer_1803886264724
// MI455X (gfx1250) — matrix operand order repaired and checked by reading, not compiled, not run on hardware
//
#include <hip/hip_runtime.h>

constexpr int Bc = 8;
constexpr int Nc = 512;
constexpr int Dc = 128;
constexpr int Hc = 8;
constexpr int HDc = Hc * Dc;   // 1024
constexpr int NBONDc = 8;

typedef __attribute__((ext_vector_type(16))) __bf16 v16bf;
typedef __attribute__((ext_vector_type(8)))  __bf16 v8bf;
typedef __attribute__((ext_vector_type(8)))  float  v8f;

// ---- WMMA helpers (wave32; A and B share a contiguous k order) -------------------
// A (16x32 bf16): lane L -> row M=L&15, half=L>>4; vector elems e:
//   K = 16*half + e  -> two contiguous 8-elem (16B) groups.
__device__ __forceinline__ v16bf ldA(const __bf16* base, int ldk, int kb, int lane) {
  int m = lane & 15, hf = lane >> 4;
  const __bf16* p = base + (size_t)m * ldk + kb + 16 * hf;
  v8bf lo = *(const v8bf*)(p);
  v8bf hi = *(const v8bf*)(p + 8);
  return __builtin_shufflevector(lo, hi, 0,1,2,3,4,5,6,7,8,9,10,11,12,13,14,15);
}
// B (32x16 bf16): lane L -> col N=L&15; K = 16*(L>>4) + e, e=0..15 contiguous.
// colbase must point at storage where B's column n is a contiguous K-run.
__device__ __forceinline__ v16bf ldB(const __bf16* colbase, int ldk, int kb, int lane) {
  int n = lane & 15, hf = lane >> 4;
  return *(const v16bf*)(colbase + (size_t)n * ldk + kb + 16 * hf);
}
__device__ __forceinline__ v8f mma_bf16(v16bf a, v16bf b, v8f c) {
  return __builtin_amdgcn_wmma_f32_16x16x32_bf16(false, a, false, b, (short)0, c, false, false);
}

// ---- K0: weight transpose->bf16, bond row-sums ----------------------------
__global__ void __launch_bounds__(256) prep_kernel(
    const float* __restrict__ Wq, const float* __restrict__ Wk, const float* __restrict__ Wv,
    const float* __restrict__ outW, const float* __restrict__ bond_emb,
    __bf16* __restrict__ WqT, __bf16* __restrict__ WkT, __bf16* __restrict__ WvT,
    __bf16* __restrict__ outWT, float* __restrict__ wsum) {
  int i = blockIdx.x * 256 + threadIdx.x;
  if (i < Hc * Dc * Dc) {             // == Dc*HDc == 131072, covers both transposes
    int h = i >> 14, r = i & 16383, e = r >> 7, d = r & 127;
    int s = h * 16384 + d * 128 + e;  // W[h][d][e] -> WT[h][e][d]
    WqT[i] = (__bf16)Wq[s];
    WkT[i] = (__bf16)Wk[s];
    WvT[i] = (__bf16)Wv[s];
    int d2 = i >> 10, c = i & 1023;   // outW[c][d] -> outWT[d][c]
    outWT[i] = (__bf16)outW[c * Dc + d2];
  }
  if (i < NBONDc) {
    float s = 0.f;
    for (int d = 0; d < Dc; ++d) s += bond_emb[i * Dc + d];
    wsum[i] = s;
  }
}

// ---- K1: t = relu(ff(concat(elem_emb[TE], relu(TP@posW+pb)))) -> bf16 -----
__global__ void __launch_bounds__(128) node_kernel(
    const int* __restrict__ TE, const float* __restrict__ TP,
    const float* __restrict__ elem_emb, const float* __restrict__ pos_W,
    const float* __restrict__ pos_b, const float* __restrict__ ff_W,
    const float* __restrict__ ff_b, __bf16* __restrict__ t_bf) {
  __shared__ float cat[2 * Dc];
  int bn = blockIdx.x, d = threadIdx.x;
  float te = elem_emb[TE[bn] * Dc + d];
  float x0 = TP[bn * 3 + 0], x1 = TP[bn * 3 + 1], x2 = TP[bn * 3 + 2];
  float tp = fmaf(x0, pos_W[d], fmaf(x1, pos_W[Dc + d], fmaf(x2, pos_W[2 * Dc + d], pos_b[d])));
  cat[d] = te;
  cat[Dc + d] = fmaxf(tp, 0.f);
  __syncthreads();
  float s = ff_b[d];
  #pragma unroll 8
  for (int c = 0; c < 2 * Dc; ++c) s = fmaf(cat[c], ff_W[c * Dc + d], s);
  t_bf[(size_t)bn * Dc + d] = (__bf16)fmaxf(s, 0.f);
}

// ---- K2: q/k/v = t @ W + b (one wave per (b,h,which,16-row tile)) ---------
__global__ void __launch_bounds__(32) qkv_kernel(
    const __bf16* __restrict__ t_bf,
    const __bf16* __restrict__ WqT, const __bf16* __restrict__ WkT, const __bf16* __restrict__ WvT,
    const float* __restrict__ bq, const float* __restrict__ bk, const float* __restrict__ bv,
    __bf16* __restrict__ qO, __bf16* __restrict__ kO, __bf16* __restrict__ vTO) {
  int bx = blockIdx.x;
  int mtile = bx & 31; int t2 = bx >> 5;
  int which = t2 % 3;  int t3 = t2 / 3;
  int h = t3 & 7;      int b = t3 >> 3;
  int lane = threadIdx.x;
  int nn = lane & 15, hf = lane >> 4, Mb = 8 * hf;

  const __bf16* A = t_bf + ((size_t)b * Nc + mtile * 16) * Dc;
  const __bf16* W = (which == 0 ? WqT : which == 1 ? WkT : WvT) + (size_t)h * Dc * Dc;
  const float* bias = (which == 0 ? bq : which == 1 ? bk : bv) + h * Dc;

  v16bf a[4];
  #pragma unroll
  for (int kc = 0; kc < 4; ++kc) a[kc] = ldA(A, Dc, kc * 32, lane);

  for (int nt = 0; nt < 8; ++nt) {
    float bb = bias[nt * 16 + nn];
    v8f acc;
    #pragma unroll
    for (int g = 0; g < 8; ++g) acc[g] = bb;
    #pragma unroll
    for (int kc = 0; kc < 4; ++kc) {
      v16bf bm = ldB(W + (size_t)nt * 16 * Dc, Dc, kc * 32, lane);
      acc = mma_bf16(a[kc], bm, acc);
    }
    if (which < 2) {  // q,k row-major [b,h,n,d]
      __bf16* dst = (which == 0 ? qO : kO) +
                    ((size_t)(b * Hc + h) * Nc + mtile * 16) * Dc + nt * 16 + nn;
      #pragma unroll
      for (int g = 0; g < 8; ++g) dst[(size_t)(Mb + g) * Dc] = (__bf16)acc[g];
    } else {          // v transposed [b,h,feat,keypos] -> contiguous over M
      v8bf pk;
      #pragma unroll
      for (int g = 0; g < 8; ++g) pk[g] = (__bf16)acc[g];
      *(v8bf*)(vTO + ((size_t)(b * Hc + h) * Dc + nt * 16 + nn) * Nc + mtile * 16 + Mb) = pk;
    }
  }
}

// ---- K3: causal attention with multiplicative edge weights ----------------
// LDS: a single fp32 score buffer (33KB); the bf16 probability matrix is
// packed IN PLACE over the score rows (byte 2j of a row overlays float j/2,
// and j only ascends, so every float is consumed before it is clobbered).
__global__ void __launch_bounds__(32) attn_kernel(
    const __bf16* __restrict__ q, const __bf16* __restrict__ k, const __bf16* __restrict__ vT,
    const int* __restrict__ TADJ, const float* __restrict__ wsum,
    __bf16* __restrict__ o) {
  constexpr int SST = Nc + 4;   // float stride: 516 (2064B, 16B-aligned, no 64-bank conflicts)
  __shared__ float sc[16][SST];

  int bx = blockIdx.x;
  int mtile = bx & 31; int h = (bx >> 5) & 7; int b = bx >> 8;
  int lane = threadIdx.x;
  int nn = lane & 15, hf = lane >> 4, Mb = 8 * hf;
  const float scale = 0.04419417382415922f;  // 1/sqrt(512)

  const __bf16* qp = q + ((size_t)(b * Hc + h) * Nc + mtile * 16) * Dc;
  const __bf16* kp = k + (size_t)(b * Hc + h) * Nc * Dc;

  v16bf a[4];
  #pragma unroll
  for (int kc = 0; kc < 4; ++kc) a[kc] = ldA(qp, Dc, kc * 32, lane);

  for (int jt = 0; jt <= mtile; ++jt) {       // causal: only lower tiles
    if (jt < mtile) {                         // prefetch next 4KB k-tile (gfx1250 path)
      const __bf16* nxt = kp + (size_t)(jt + 1) * 16 * Dc;
      __builtin_prefetch(nxt + lane * 64, 0, 1);   // 32 lanes x 128B cachelines
    }
    v8f s = {};
    #pragma unroll
    for (int kc = 0; kc < 4; ++kc) {
      // B = k^T: column n of B is k row n -> contiguous
      v16bf bm = ldB(kp + (size_t)jt * 16 * Dc, Dc, kc * 32, lane);
      s = mma_bf16(a[kc], bm, s);
    }
    int kcol = jt * 16 + nn;
    #pragma unroll
    for (int g = 0; g < 8; ++g) {
      int mrow = mtile * 16 + Mb + g;
      float val = s[g] * scale;
      val *= wsum[TADJ[((size_t)b * Nc + mrow) * Nc + kcol]];
      if (kcol > mrow) val = -__builtin_inff();
      sc[Mb + g][kcol] = val;
    }
  }
  __syncthreads();

  int len = (mtile + 1) * 16;
  int lenpad = (len + 31) & ~31;
  {
    // all 32 lanes: two lanes per row (stride-2 columns), combine via lane^16
    int r = nn;
    float mx = -__builtin_inff();
    for (int j = hf; j < len; j += 2) mx = fmaxf(mx, sc[r][j]);
    mx = fmaxf(mx, __shfl_xor(mx, 16, 32));
    float sum = 0.f;
    for (int j = hf; j < len; j += 2) { float e = __expf(sc[r][j] - mx); sum += e; sc[r][j] = e; }
    sum += __shfl_xor(sum, 16, 32);
    float inv = 1.f / sum;
    __bf16* pbr = (__bf16*)&sc[r][0];          // in-place bf16 overlay of this row
    for (int j = hf; j < len; j += 2) pbr[j] = (__bf16)(sc[r][j] * inv);
    for (int j = len + hf; j < lenpad; j += 2) pbr[j] = (__bf16)0.f;
  }
  __syncthreads();

  const __bf16* vp = vT + (size_t)(b * Hc + h) * Dc * Nc;
  int nchunk = lenpad >> 5;
  for (int nt = 0; nt < 8; ++nt) {
    v8f acc = {};
    for (int c = 0; c < nchunk; ++c) {
      // P from the bf16 overlay: row stride = 2*SST bf16 elements
      v16bf pa = ldA((const __bf16*)&sc[0][0], 2 * SST, c * 32, lane);
      v16bf bm = ldB(vp + (size_t)nt * 16 * Nc, Nc, c * 32, lane);  // v^T cols contiguous
      acc = mma_bf16(pa, bm, acc);
    }
    __bf16* dst = o + ((size_t)b * Nc + mtile * 16) * HDc + h * Dc + nt * 16 + nn;
    #pragma unroll
    for (int g = 0; g < 8; ++g) dst[(size_t)(Mb + g) * HDc] = (__bf16)acc[g];
  }
}

// ---- K4: out = o[16x1024] @ outW[1024x128] + out_b ------------------------
__global__ void __launch_bounds__(32) outproj_kernel(
    const __bf16* __restrict__ o, const __bf16* __restrict__ outWT,
    const float* __restrict__ outb, float* __restrict__ out) {
  int rowbase = blockIdx.x * 16;   // over B*N
  int lane = threadIdx.x;
  int nn = lane & 15, hf = lane >> 4, Mb = 8 * hf;

  v8f acc[8];
  #pragma unroll
  for (int nt = 0; nt < 8; ++nt) {
    float bb = outb[nt * 16 + nn];
    #pragma unroll
    for (int g = 0; g < 8; ++g) acc[nt][g] = bb;
  }
  for (int kc = 0; kc < 32; ++kc) {
    v16bf a = ldA(o + (size_t)rowbase * HDc, HDc, kc * 32, lane);
    #pragma unroll
    for (int nt = 0; nt < 8; ++nt) {
      v16bf bm = ldB(outWT + (size_t)nt * 16 * HDc, HDc, kc * 32, lane);
      acc[nt] = mma_bf16(a, bm, acc[nt]);
    }
  }
  #pragma unroll
  for (int nt = 0; nt < 8; ++nt) {
    #pragma unroll
    for (int g = 0; g < 8; ++g)
      out[(size_t)(rowbase + Mb + g) * Dc + nt * 16 + nn] = acc[nt][g];
  }
}

extern "C" void kernel_launch(void* const* d_in, const int* in_sizes, int n_in,
                              void* d_out, int out_size, void* d_ws, size_t ws_size,
                              hipStream_t stream) {
  const int*   TE       = (const int*)d_in[0];
  const float* TP       = (const float*)d_in[1];
  const int*   TADJ     = (const int*)d_in[2];
  const float* elem_emb = (const float*)d_in[3];
  const float* bond_emb = (const float*)d_in[4];
  const float* pos_W    = (const float*)d_in[5];
  const float* pos_b    = (const float*)d_in[6];
  const float* ff_W     = (const float*)d_in[7];
  const float* ff_b     = (const float*)d_in[8];
  const float* Wq       = (const float*)d_in[9];
  const float* bq       = (const float*)d_in[10];
  const float* Wk       = (const float*)d_in[11];
  const float* bk       = (const float*)d_in[12];
  const float* Wv       = (const float*)d_in[13];
  const float* bv       = (const float*)d_in[14];
  const float* outW     = (const float*)d_in[15];
  const float* outb     = (const float*)d_in[16];
  float* out = (float*)d_out;
  (void)in_sizes; (void)n_in; (void)out_size; (void)ws_size;

  // workspace bump allocator (256B aligned); total ~35 MB
  size_t off = 0;
  char* wsb = (char*)d_ws;
  auto take = [&](size_t bytes) -> void* {
    void* p = wsb + off;
    off += (bytes + 255) & ~(size_t)255;
    return p;
  };
  __bf16* WqT   = (__bf16*)take((size_t)Hc * Dc * Dc * 2);
  __bf16* WkT   = (__bf16*)take((size_t)Hc * Dc * Dc * 2);
  __bf16* WvT   = (__bf16*)take((size_t)Hc * Dc * Dc * 2);
  __bf16* outWT = (__bf16*)take((size_t)Dc * HDc * 2);
  float*  wsum  = (float*) take((size_t)NBONDc * 4);
  __bf16* t_bf  = (__bf16*)take((size_t)Bc * Nc * Dc * 2);
  __bf16* d_q   = (__bf16*)take((size_t)Bc * Hc * Nc * Dc * 2);
  __bf16* d_k   = (__bf16*)take((size_t)Bc * Hc * Nc * Dc * 2);
  __bf16* d_vT  = (__bf16*)take((size_t)Bc * Hc * Nc * Dc * 2);
  __bf16* d_o   = (__bf16*)take((size_t)Bc * Nc * HDc * 2);

  prep_kernel<<<(Hc * Dc * Dc + 255) / 256, 256, 0, stream>>>(
      Wq, Wk, Wv, outW, bond_emb, WqT, WkT, WvT, outWT, wsum);

  node_kernel<<<Bc * Nc, Dc, 0, stream>>>(
      TE, TP, elem_emb, pos_W, pos_b, ff_W, ff_b, t_bf);

  qkv_kernel<<<Bc * Hc * 3 * (Nc / 16), 32, 0, stream>>>(
      t_bf, WqT, WkT, WvT, bq, bk, bv, d_q, d_k, d_vT);

  attn_kernel<<<Bc * Hc * (Nc / 16), 32, 0, stream>>>(
      d_q, d_k, d_vT, TADJ, wsum, d_o);

  outproj_kernel<<<Bc * Nc / 16, 32, 0, stream>>>(
      d_o, outWT, outb, out);
}
